// Decoder_81638738362726
// MI455X (gfx1250) — matrix operand order repaired and checked by reading, not compiled, not run on hardware
//
#include <hip/hip_runtime.h>
#include <hip/hip_bf16.h>
#include <float.h>

// Problem constants (match reference)
#define SLOPE 0.1f
#define Bn   32
#define Tn   128
#define En   512      // EMBED
#define Hn   1024     // DEC
#define C2n  1024     // 2*ENC
#define Vn   32000    // VOCAB
#define Gn   3072     // 3*DEC
#define Xn   1536     // EMBED + 2*ENC

typedef __bf16 bf16_t;
typedef __attribute__((ext_vector_type(16))) __bf16 v16bf;
typedef __attribute__((ext_vector_type(8)))  __bf16 v8bf;
typedef __attribute__((ext_vector_type(8)))  float  v8f;

__device__ __forceinline__ unsigned short f2bf(float f) {
  union { float f; unsigned u; } c; c.f = f;
  unsigned u = c.u;
  u += 0x7FFFu + ((u >> 16) & 1u);   // round-to-nearest-even
  return (unsigned short)(u >> 16);
}

// ---------------- setup kernels ----------------

__global__ void cvt_f32_bf16(const float* __restrict__ src,
                             unsigned short* __restrict__ dst, int n) {
  int i = blockIdx.x * blockDim.x + threadIdx.x;
  if (i < n) dst[i] = f2bf(src[i]);
}

__global__ void embed_mask(const int* __restrict__ tokens,
                           const int* __restrict__ lengths,
                           const float* __restrict__ emb,
                           unsigned short* __restrict__ padded) {
  int bt = blockIdx.x;              // [0, B*T)
  int b = bt / Tn, t = bt % Tn;
  int tok = tokens[bt];
  float m = (t < lengths[b]) ? 1.0f : 0.0f;
  const float* e = emb + (long)tok * En;
  for (int i = threadIdx.x; i < En; i += blockDim.x)
    padded[(long)bt * En + i] = f2bf(e[i] * m);
}

__global__ void init_h(const float* __restrict__ hidden,
                       float* __restrict__ h,
                       unsigned short* __restrict__ h_bf) {
  int i = blockIdx.x * blockDim.x + threadIdx.x;  // Bn*Hn threads
  float v = hidden[i];
  h[i] = v;
  h_bf[i] = f2bf(v);
}

__global__ void init_out0(float* __restrict__ out) {
  int i = blockIdx.x * blockDim.x + threadIdx.x;  // Bn*Vn threads
  int b = i / Vn, v = i % Vn;
  out[(long)b * Tn * Vn + v] = (v == 0) ? 1.0f : 0.0f;
}

// ---------------- WMMA bf16 GEMM ----------------
// C[M,N] = A[M,K](bf16,row-major,lda) * W[N,K]^T(bf16,row-major) + bias
// One wave per 16(N) x 32(M) output block: TWO 16x16 WMMA tiles sharing a
// single B fragment per K-step -> halves traffic on the streaming W operand
// (W_out is the dominant stream, L2-resident in bf16).
// M multiple of 32, N multiple of 16, K multiple of 32. act!=0 -> LeakyReLU(SLOPE).
__global__ __launch_bounds__(32)
void gemm_bf16_wmma(const bf16_t* __restrict__ A, int lda,
                    const bf16_t* __restrict__ W, int K,
                    const float* __restrict__ bias,
                    float* __restrict__ Cout, long ldc, int act) {
  const int n0   = blockIdx.x * 16;
  const int m0   = blockIdx.y * 32;
  const int lane = threadIdx.x;
  const int lo16 = lane & 15;
  const int half = lane >> 4;              // 0 or 1

  v8f acc0 = {};
  v8f acc1 = {};
  // A: lane lo16 = row M; half selects K half (16 contiguous)
  const bf16_t* arow0 = A + (long)(m0 + lo16) * lda + half * 16;
  const bf16_t* arow1 = arow0 + (long)16 * lda;
  // B: lane lo16 = column N (row of W); half selects K half (16 contiguous)
  const bf16_t* wrow  = W + (long)(n0 + lo16) * K + half * 16;

  for (int k = 0; k < K; k += 32) {
    __builtin_prefetch(wrow + k + 128, 0, 0);     // global_prefetch_b8
    v16bf bfrag = *(const v16bf*)(wrow + k);      // shared by both M-tiles

    v8bf a0lo = *(const v8bf*)(arow0 + k);
    v8bf a0hi = *(const v8bf*)(arow0 + k + 8);
    v16bf a0;
#pragma unroll
    for (int i = 0; i < 8; ++i) { a0[i] = a0lo[i]; a0[i + 8] = a0hi[i]; }
    acc0 = __builtin_amdgcn_wmma_f32_16x16x32_bf16(
        false, a0, false, bfrag, (short)0, acc0, false, false);

    v8bf a1lo = *(const v8bf*)(arow1 + k);
    v8bf a1hi = *(const v8bf*)(arow1 + k + 8);
    v16bf a1;
#pragma unroll
    for (int i = 0; i < 8; ++i) { a1[i] = a1lo[i]; a1[i + 8] = a1hi[i]; }
    acc1 = __builtin_amdgcn_wmma_f32_16x16x32_bf16(
        false, a1, false, bfrag, (short)0, acc1, false, false);
  }

  const int col = n0 + lo16;
  const float bv = bias ? bias[col] : 0.0f;
#pragma unroll
  for (int r = 0; r < 8; ++r) {
    int row = m0 + r + 8 * half;               // C/D layout per ISA
    float v0 = acc0[r] + bv;
    float v1 = acc1[r] + bv;
    if (act) {
      v0 = (v0 > 0.0f) ? v0 : SLOPE * v0;
      v1 = (v1 > 0.0f) ? v1 : SLOPE * v1;
    }
    Cout[(long)row * ldc + col] = v0;
    Cout[(long)(row + 16) * ldc + col] = v1;
  }
}

// ---------------- attention: scores -> softmax -> context -> x ----------------
// One block per batch row b. 128 threads (4 waves).
__global__ __launch_bounds__(128)
void attn_step(const float* __restrict__ enc_proj,   // [B,T,H]
               const float* __restrict__ dec_proj,   // [B,H]
               const float* __restrict__ v_a,        // [H]
               const float* __restrict__ enc_outs,   // [B,T,C2] f32
               const int* __restrict__ lengths,
               const unsigned short* __restrict__ padded, // [B,T,E] bf16
               unsigned short* __restrict__ x_bf,    // [B,Xn] bf16
               int tstep) {
  __shared__ float s_sc[Tn];
  __shared__ float s_red[4];
  const int b = blockIdx.x;
  const int tid = threadIdx.x;
  const int wave = tid >> 5, lane = tid & 31;
  const int len = lengths[b];
  const float* dp = dec_proj + (long)b * Hn;

  // additive-attention scores, one wave per t
  for (int t = wave; t < Tn; t += 4) {
    const float* ep = enc_proj + ((long)b * Tn + t) * Hn;
    float p = 0.0f;
    for (int a = lane; a < Hn; a += 32)
      p += v_a[a] * tanhf(ep[a] + dp[a]);
#pragma unroll
    for (int off = 16; off; off >>= 1) p += __shfl_xor(p, off, 32);
    if (lane == 0) s_sc[t] = (t < len) ? p : -FLT_MAX;
  }
  __syncthreads();

  // softmax over T=128 (one score per thread)
  float mx = s_sc[tid];
#pragma unroll
  for (int off = 16; off; off >>= 1) mx = fmaxf(mx, __shfl_xor(mx, off, 32));
  if (lane == 0) s_red[wave] = mx;
  __syncthreads();
  mx = fmaxf(fmaxf(s_red[0], s_red[1]), fmaxf(s_red[2], s_red[3]));
  float e = expf(s_sc[tid] - mx);
  __syncthreads();                              // done reading s_red
  float sm = e;
#pragma unroll
  for (int off = 16; off; off >>= 1) sm += __shfl_xor(sm, off, 32);
  if (lane == 0) s_red[wave] = sm;
  __syncthreads();
  sm = s_red[0] + s_red[1] + s_red[2] + s_red[3];
  s_sc[tid] = e / sm;                           // attention weights
  __syncthreads();

  // context[b,c] = sum_t attn[t] * enc_outs[b,t,c]  -> x_bf[b, En + c]
  const float* eo = enc_outs + (long)b * Tn * C2n;
  for (int c = tid; c < C2n; c += 128) {
    float acc = 0.0f;
    for (int t = 0; t < Tn; ++t) acc += s_sc[t] * eo[(long)t * C2n + c];
    x_bf[(long)b * Xn + En + c] = f2bf(acc);
  }
  // teacher-forced input: x_bf[b, 0..En) = padded[b, tstep-1, :]
  const unsigned short* sel = padded + ((long)b * Tn + (tstep - 1)) * En;
  for (int i = tid; i < En; i += 128)
    x_bf[(long)b * Xn + i] = sel[i];
}

// ---------------- GRU gate elementwise ----------------
__global__ __launch_bounds__(256)
void gru_gate(const float* __restrict__ gi, const float* __restrict__ gh,
              float* __restrict__ h, unsigned short* __restrict__ h_bf) {
  const int b = blockIdx.x;
  const long base = (long)b * Gn;
  for (int n = threadIdx.x; n < Hn; n += blockDim.x) {
    float r  = 1.0f / (1.0f + expf(-(gi[base + n]          + gh[base + n])));
    float z  = 1.0f / (1.0f + expf(-(gi[base + Hn + n]     + gh[base + Hn + n])));
    float nn = tanhf(gi[base + 2 * Hn + n] + r * gh[base + 2 * Hn + n]);
    long hi = (long)b * Hn + n;
    float hv = (1.0f - z) * nn + z * h[hi];
    h[hi] = hv;
    h_bf[hi] = f2bf(hv);
  }
}

// ---------------- host orchestration ----------------
extern "C" void kernel_launch(void* const* d_in, const int* in_sizes, int n_in,
                              void* d_out, int out_size, void* d_ws, size_t ws_size,
                              hipStream_t stream) {
  const int*   tokens   = (const int*)  d_in[0];
  const int*   lengths  = (const int*)  d_in[1];
  const float* hidden   = (const float*)d_in[2];
  const float* enc_outs = (const float*)d_in[3];
  const float* emb      = (const float*)d_in[4];
  const float* W_ih     = (const float*)d_in[5];
  const float* b_ih     = (const float*)d_in[6];
  const float* W_hh     = (const float*)d_in[7];
  const float* b_hh     = (const float*)d_in[8];
  const float* Wa_enc   = (const float*)d_in[9];
  const float* Wa_dec   = (const float*)d_in[10];
  const float* v_a      = (const float*)d_in[11];
  const float* W_out    = (const float*)d_in[12];
  const float* b_out    = (const float*)d_in[13];
  float* out = (float*)d_out;

  // workspace carve-up (256B aligned)
  char* ws = (char*)d_ws;
  size_t off = 0;
  auto carve = [&](size_t bytes) -> char* {
    char* p = ws + off;
    off = (off + bytes + 255) & ~(size_t)255;
    return p;
  };
  unsigned short* W_out_bf  = (unsigned short*)carve((size_t)Vn * Hn * 2);   // 64 MB (L2-resident)
  unsigned short* W_ih_bf   = (unsigned short*)carve((size_t)Gn * Xn * 2);
  unsigned short* W_hh_bf   = (unsigned short*)carve((size_t)Gn * Hn * 2);
  unsigned short* Wa_enc_bf = (unsigned short*)carve((size_t)Hn * C2n * 2);
  unsigned short* Wa_dec_bf = (unsigned short*)carve((size_t)Hn * Hn * 2);
  unsigned short* enc_bf    = (unsigned short*)carve((size_t)Bn * Tn * C2n * 2);
  float*          enc_proj  = (float*)         carve((size_t)Bn * Tn * Hn * 4);
  unsigned short* padded    = (unsigned short*)carve((size_t)Bn * Tn * En * 2);
  float*          h         = (float*)         carve((size_t)Bn * Hn * 4);
  unsigned short* h_bf      = (unsigned short*)carve((size_t)Bn * Hn * 2);
  float*          dec_proj  = (float*)         carve((size_t)Bn * Hn * 4);
  unsigned short* x_bf      = (unsigned short*)carve((size_t)Bn * Xn * 2);
  float*          gi        = (float*)         carve((size_t)Bn * Gn * 4);
  float*          gh        = (float*)         carve((size_t)Bn * Gn * 4);
  (void)ws_size; (void)n_in; (void)in_sizes; (void)out_size;

  auto cvt = [&](const float* s, unsigned short* d, long n) {
    cvt_f32_bf16<<<(unsigned)((n + 255) / 256), 256, 0, stream>>>(s, d, (int)n);
  };
  // one-time bf16 conversions
  cvt(W_out,  W_out_bf,  (long)Vn * Hn);
  cvt(W_ih,   W_ih_bf,   (long)Gn * Xn);
  cvt(W_hh,   W_hh_bf,   (long)Gn * Hn);
  cvt(Wa_enc, Wa_enc_bf, (long)Hn * C2n);
  cvt(Wa_dec, Wa_dec_bf, (long)Hn * Hn);
  cvt(enc_outs, enc_bf,  (long)Bn * Tn * C2n);

  embed_mask<<<Bn * Tn, 256, 0, stream>>>(tokens, lengths, emb, padded);
  init_h<<<(Bn * Hn) / 256, 256, 0, stream>>>(hidden, h, h_bf);
  init_out0<<<(Bn * Vn) / 256, 256, 0, stream>>>(out);

  // enc_proj[b,t,:] = encoder_outs[b,t,:] @ Wa_enc^T   (M=B*T=4096, N=1024, K=1024)
  gemm_bf16_wmma<<<dim3(Hn / 16, (Bn * Tn) / 32), 32, 0, stream>>>(
      (const bf16_t*)enc_bf, C2n, (const bf16_t*)Wa_enc_bf, C2n,
      nullptr, enc_proj, (long)Hn, 0);

  // sequential decode: t = 1 .. T-1
  for (int t = 1; t < Tn; ++t) {
    // dec_proj = h @ Wa_dec^T  (32 x 1024 x 1024)
    gemm_bf16_wmma<<<dim3(Hn / 16, Bn / 32), 32, 0, stream>>>(
        (const bf16_t*)h_bf, Hn, (const bf16_t*)Wa_dec_bf, Hn,
        nullptr, dec_proj, (long)Hn, 0);
    // attention -> x_bf = [sel_{t-1}, context]
    attn_step<<<Bn, 128, 0, stream>>>(enc_proj, dec_proj, v_a, enc_outs,
                                      lengths, padded, x_bf, t);
    // gi = x @ W_ih^T + b_ih  (32 x 3072 x 1536)
    gemm_bf16_wmma<<<dim3(Gn / 16, Bn / 32), 32, 0, stream>>>(
        (const bf16_t*)x_bf, Xn, (const bf16_t*)W_ih_bf, Xn,
        b_ih, gi, (long)Gn, 0);
    // gh = h @ W_hh^T + b_hh  (32 x 3072 x 1024)
    gemm_bf16_wmma<<<dim3(Gn / 16, Bn / 32), 32, 0, stream>>>(
        (const bf16_t*)h_bf, Hn, (const bf16_t*)W_hh_bf, Hn,
        b_hh, gh, (long)Gn, 0);
    // gate math, h update (in place) + bf16 shadow
    gru_gate<<<Bn, 256, 0, stream>>>(gi, gh, h, h_bf);
    // logits = LeakyReLU(h_new @ W_out^T + b_out) -> out[:, t, :]
    gemm_bf16_wmma<<<dim3(Vn / 16, Bn / 32), 32, 0, stream>>>(
        (const bf16_t*)h_bf, Hn, (const bf16_t*)W_out_bf, Hn,
        b_out, out + (long)t * Vn, (long)Tn * Vn, 1);
  }
}
